// PureTriXButterfly_63806034149897
// MI455X (gfx1250) — matrix operand order repaired and checked by reading, not compiled, not run on hardware
//
#include <hip/hip_runtime.h>
#include <hip/hip_bf16.h>
#include <math.h>

// ---------------------------------------------------------------------------
// MI455X (gfx1250, wave32) implementation.
//  - bf16 WMMA (v_wmma_f32_16x16x32_bf16) for router layer-1 GEMM and routed
//    expert FF1 GEMM.
//  - FF2 algebraically collapsed into per-tile vectors v_sum/v_diff.
//  - Top-1 routed dispatch (per-tile gathered token lists).
//  - Double-buffered LDS tiles; CDNA5 async global->LDS copies (ASYNCcnt).
// ---------------------------------------------------------------------------

typedef __attribute__((ext_vector_type(16))) __bf16  v16bf;
typedef __attribute__((ext_vector_type(8)))  float   v8f;
typedef __attribute__((ext_vector_type(4)))  unsigned int u32x4;

union Frag16 { v16bf bf; u32x4 u[2]; };

#define SUMF  (1u << 24)
#define DIFFF (1u << 25)
#define TOKM  0x00FFFFFFu

#if __has_builtin(__builtin_amdgcn_global_load_async_to_lds_b128)
#define HAVE_ASYNC_LDS 1
#else
#define HAVE_ASYNC_LDS 0
#endif

// 16-byte global -> LDS copy (per lane). Async (ASYNCcnt-tracked) on CDNA5.
__device__ __forceinline__ void cp16(const unsigned short* g, unsigned short* l) {
#if HAVE_ASYNC_LDS
    typedef int vec4i __attribute__((__vector_size__(4 * sizeof(int))));
    typedef __attribute__((address_space(1))) vec4i* gp_t;   // global
    typedef __attribute__((address_space(3))) vec4i* lp_t;   // LDS
    __builtin_amdgcn_global_load_async_to_lds_b128((gp_t)g, (lp_t)l, 0, 0);
#else
    *reinterpret_cast<u32x4*>(l) = *reinterpret_cast<const u32x4*>(g);
#endif
}
__device__ __forceinline__ void cp_commit() {
#if HAVE_ASYNC_LDS
#if __has_builtin(__builtin_amdgcn_s_wait_asynccnt)
    __builtin_amdgcn_s_wait_asynccnt(0);
#else
    asm volatile("s_wait_asynccnt 0x0" ::: "memory");
#endif
#endif
}

__device__ __forceinline__ unsigned short f2bf(float f) {
    unsigned int u = __float_as_uint(f);
    u += 0x7FFFu + ((u >> 16) & 1u);          // round-to-nearest-even
    return (unsigned short)(u >> 16);
}
__device__ __forceinline__ float bf2f(unsigned short h) {
    return __uint_as_float(((unsigned int)h) << 16);
}
__device__ __forceinline__ float gelu_exact(float x) {
    return 0.5f * x * (1.0f + erff(x * 0.70710678118654752440f));
}

// ---------------------------------------------------------------------------
// f32 -> bf16 weight conversion (4 elems / thread)
// ---------------------------------------------------------------------------
__global__ void cvt_bf16_kernel(const float* __restrict__ in,
                                unsigned short* __restrict__ out, int n) {
    int i = (blockIdx.x * 256 + threadIdx.x) * 4;
    if (i >= n) return;
    out[i + 0] = f2bf(in[i + 0]);
    out[i + 1] = f2bf(in[i + 1]);
    out[i + 2] = f2bf(in[i + 2]);
    out[i + 3] = f2bf(in[i + 3]);
}

// ---------------------------------------------------------------------------
// v_sum[t][k] = dot(We2[t][k][:], w_sum); v_diff likewise. One wave per row.
// ---------------------------------------------------------------------------
__global__ __launch_bounds__(256) void vproj_kernel(
    const float* __restrict__ We2, const float* __restrict__ wsum,
    const float* __restrict__ wdiff, float* __restrict__ vs,
    float* __restrict__ vd) {
    int wv = threadIdx.x >> 5, lane = threadIdx.x & 31;
    int row = blockIdx.x * 8 + wv;               // 0..16383
    const float* r = We2 + (size_t)row * 1024;
    float s = 0.f, d = 0.f;
    for (int k = lane; k < 1024; k += 32) {
        float w = r[k];
        s += w * wsum[k];
        d += w * wdiff[k];
    }
    for (int o = 16; o > 0; o >>= 1) {
        s += __shfl_xor(s, o, 32);
        d += __shfl_xor(d, o, 32);
    }
    if (lane == 0) { vs[row] = s; vd[row] = d; }
}

// c_sum[t] = dot(be2[t], w_sum); c_diff[t] = dot(be2[t], w_diff)
__global__ void cvec_kernel(const float* __restrict__ be2,
                            const float* __restrict__ wsum,
                            const float* __restrict__ wdiff,
                            float* __restrict__ cs, float* __restrict__ cd) {
    int t = blockIdx.x, tid = threadIdx.x;
    float s = 0.f, d = 0.f;
    for (int k = tid; k < 1024; k += 256) {
        float b = be2[(size_t)t * 1024 + k];
        s += b * wsum[k];
        d += b * wdiff[k];
    }
    int lane = tid & 31, wv = tid >> 5;
    for (int o = 16; o > 0; o >>= 1) {
        s += __shfl_xor(s, o, 32);
        d += __shfl_xor(d, o, 32);
    }
    __shared__ float shs[8], shd[8];
    if (lane == 0) { shs[wv] = s; shd[wv] = d; }
    __syncthreads();
    if (tid == 0) {
        float S = 0.f, D = 0.f;
        for (int i = 0; i < 8; ++i) { S += shs[i]; D += shd[i]; }
        cs[t] = S; cd[t] = D;
    }
}

// ---------------------------------------------------------------------------
// Encoder: fourier feats (24) -> x = gelu(LN(feat@W_in + b_in)) -> bf16.
// One block (256 thr) per token; feats computed once into LDS.
// ---------------------------------------------------------------------------
__global__ __launch_bounds__(256) void encode_kernel(
    const float* __restrict__ a, const float* __restrict__ b,
    const float* __restrict__ Win, const float* __restrict__ bin,
    const float* __restrict__ lng, const float* __restrict__ lnb,
    unsigned short* __restrict__ X) {
    const int tok = blockIdx.x;
    const int tid = threadIdx.x;
    __shared__ float feats[24];
    if (tid < 24) {
        int f = tid % 6, which = tid / 6;          // 0:sin_a 1:cos_a 2:sin_b 3:cos_b
        float val = (which < 2) ? a[tok] : b[tok];
        float ang = val * 0.39269908169872414f * (float)(1 << f);  // 2pi/16 * 2^f
        feats[tid] = (which & 1) ? cosf(ang) : sinf(ang);
    }
    __syncthreads();
    float xloc[4];
    float s = 0.f, q = 0.f;
    #pragma unroll
    for (int p = 0; p < 4; ++p) {
        int j = tid + 256 * p;
        float acc = bin[j];
        #pragma unroll
        for (int k = 0; k < 24; ++k) acc += feats[k] * Win[k * 1024 + j];
        xloc[p] = acc;
        s += acc;
        q += acc * acc;
    }
    int lane = tid & 31, wv = tid >> 5;
    for (int o = 16; o > 0; o >>= 1) {
        s += __shfl_xor(s, o, 32);
        q += __shfl_xor(q, o, 32);
    }
    __shared__ float rs[8], rq[8];
    if (lane == 0) { rs[wv] = s; rq[wv] = q; }
    __syncthreads();
    if (tid == 0) {
        float S = 0.f, Q = 0.f;
        for (int i = 0; i < 8; ++i) { S += rs[i]; Q += rq[i]; }
        rs[0] = S; rq[0] = Q;
    }
    __syncthreads();
    float mean = rs[0] * (1.0f / 1024.0f);
    float var  = rq[0] * (1.0f / 1024.0f) - mean * mean;
    float rstd = rsqrtf(var + 1e-5f);
    #pragma unroll
    for (int p = 0; p < 4; ++p) {
        int j = tid + 256 * p;
        float v = (xloc[p] - mean) * rstd * lng[j] + lnb[j];
        X[(size_t)tok * 1024 + j] = f2bf(gelu_exact(v));
    }
}

// ---------------------------------------------------------------------------
// bf16 GEMM: Out = gelu(A(MxK) @ W(KxN) + bias) -> bf16.
// Block 128x128, 8 waves (4Mx2N), wave 32x64 = 2x4 WMMA tiles, BK=32.
// Double-buffered LDS; async global->LDS when available.
// M,N,K assumed multiples of 128/128/32.
// ---------------------------------------------------------------------------
__global__ __launch_bounds__(256) void gemm_gelu_kernel(
    const unsigned short* __restrict__ A, const unsigned short* __restrict__ W,
    const float* __restrict__ bias, unsigned short* __restrict__ Out,
    int M, int N, int K) {
    const int tid  = threadIdx.x;
    const int lane = tid & 31;
    const int wv   = tid >> 5;
    const int wave_m = wv >> 1, wave_n = wv & 1;
    const int m0 = blockIdx.y * 128;
    const int n0 = blockIdx.x * 128;
    const int wm0 = wave_m * 32, wn0 = wave_n * 64;

    __shared__ __align__(16) unsigned short As[2][128][40];   // 32 + 8 pad
    __shared__ __align__(16) unsigned short Bs[2][32][136];   // 128 + 8 pad

    // per-thread 16B chunk coordinates (2 A chunks, 2 B chunks)
    const int ra0 = tid >> 2,            ka0 = (tid & 3) * 8;
    const int ra1 = (tid + 256) >> 2,    ka1 = ((tid + 256) & 3) * 8;
    const int kb0 = tid >> 4,            nb0 = (tid & 15) * 8;
    const int kb1 = (tid + 256) >> 4,    nb1 = ((tid + 256) & 15) * 8;
    const unsigned short* aP0 = A + (size_t)(m0 + ra0) * K + ka0;
    const unsigned short* aP1 = A + (size_t)(m0 + ra1) * K + ka1;
    const unsigned short* bP0 = W + (size_t)kb0 * N + n0 + nb0;
    const unsigned short* bP1 = W + (size_t)kb1 * N + n0 + nb1;

    v8f acc[2][4];
    const v8f vz = {0.f, 0.f, 0.f, 0.f, 0.f, 0.f, 0.f, 0.f};
    #pragma unroll
    for (int m = 0; m < 2; ++m)
        #pragma unroll
        for (int n = 0; n < 4; ++n) acc[m][n] = vz;

    const int mrow = wm0 + (lane & 15);
    const int kba  = (lane >> 4) * 8;

    // prologue: tile 0 into buffer 0
    cp16(aP0, &As[0][ra0][ka0]);
    cp16(aP1, &As[0][ra1][ka1]);
    cp16(bP0, &Bs[0][kb0][nb0]);
    cp16(bP1, &Bs[0][kb1][nb1]);
    cp_commit();
    __syncthreads();

    int cur = 0;
    for (int k0 = 0; k0 < K; k0 += 32) {
        const int nxt = cur ^ 1;
        if (k0 + 32 < K) {   // prefetch next K-tile into the other buffer
            cp16(aP0 + k0 + 32, &As[nxt][ra0][ka0]);
            cp16(aP1 + k0 + 32, &As[nxt][ra1][ka1]);
            cp16(bP0 + (size_t)(k0 + 32) * N, &Bs[nxt][kb0][nb0]);
            cp16(bP1 + (size_t)(k0 + 32) * N, &Bs[nxt][kb1][nb1]);
        }

        Frag16 af[2], bf[4];
        #pragma unroll
        for (int m = 0; m < 2; ++m) {
            af[m].u[0] = *reinterpret_cast<const u32x4*>(&As[cur][mrow + m * 16][kba]);
            af[m].u[1] = *reinterpret_cast<const u32x4*>(&As[cur][mrow + m * 16][kba + 16]);
        }
        #pragma unroll
        for (int n = 0; n < 4; ++n) {
            int nc = wn0 + n * 16 + (lane & 15);
            #pragma unroll
            for (int j = 0; j < 4; ++j) {
                bf[n].u[0][j] = (unsigned)Bs[cur][kba + 2 * j][nc] | ((unsigned)Bs[cur][kba + 2 * j + 1][nc] << 16);
                bf[n].u[1][j] = (unsigned)Bs[cur][kba + 16 + 2 * j][nc] | ((unsigned)Bs[cur][kba + 17 + 2 * j][nc] << 16);
            }
        }
        #pragma unroll
        for (int m = 0; m < 2; ++m)
            #pragma unroll
            for (int n = 0; n < 4; ++n)
                acc[m][n] = __builtin_amdgcn_wmma_f32_16x16x32_bf16(
                    false, af[m].bf, false, bf[n].bf, (short)0, acc[m][n],
                    false, false);

        cp_commit();          // prefetch (ASYNCcnt) done before anyone reads nxt
        __syncthreads();
        cur = nxt;
    }

    // epilogue: + bias, gelu, store bf16
    #pragma unroll
    for (int n = 0; n < 4; ++n) {
        int col = n0 + wn0 + n * 16 + (lane & 15);
        float bcol = bias[col];
        #pragma unroll
        for (int m = 0; m < 2; ++m) {
            #pragma unroll
            for (int i = 0; i < 8; ++i) {
                int row = m0 + wm0 + m * 16 + i + ((lane >> 4) << 3);
                float g = gelu_exact(acc[m][n][i] + bcol);
                Out[(size_t)row * N + col] = f2bf(g);
            }
        }
    }
}

// ---------------------------------------------------------------------------
// Router layer2 + argmax: tiles[i] = argmax_j (h[i]@Wr2[:,j] + br2[j]).
// One wave per token.
// ---------------------------------------------------------------------------
__global__ __launch_bounds__(256) void router_argmax_kernel(
    const unsigned short* __restrict__ H, const float* __restrict__ Wr2,
    const float* __restrict__ br2, int* __restrict__ tiles) {
    int lane = threadIdx.x & 31, wv = threadIdx.x >> 5;
    int tok = blockIdx.x * 8 + wv;
    float acc[8] = {0.f, 0.f, 0.f, 0.f, 0.f, 0.f, 0.f, 0.f};
    const unsigned short* hr = H + (size_t)tok * 1024;
    for (int k = lane; k < 1024; k += 32) {
        float hv = bf2f(hr[k]);
        const float* wr = Wr2 + (size_t)k * 8;
        #pragma unroll
        for (int j = 0; j < 8; ++j) acc[j] += hv * wr[j];
    }
    #pragma unroll
    for (int j = 0; j < 8; ++j)
        for (int o = 16; o > 0; o >>= 1) acc[j] += __shfl_xor(acc[j], o, 32);
    if (lane == 0) {
        int best = 0;
        float bv = acc[0] + br2[0];
        #pragma unroll
        for (int j = 1; j < 8; ++j) {
            float v = acc[j] + br2[j];
            if (v > bv) { bv = v; best = j; }     // first-max like jnp.argmax
        }
        tiles[tok] = best;
    }
}

__global__ void zero_counts_kernel(int* c) {
    if (threadIdx.x < 8) c[threadIdx.x] = 0;
}

// Build per-tile gathered token lists (order irrelevant -> deterministic out).
__global__ void dispatch_kernel(const int* __restrict__ ts,
                                const int* __restrict__ td,
                                int* __restrict__ counts,
                                unsigned int* __restrict__ lists, int Btok) {
    int i = blockIdx.x * 256 + threadIdx.x;
    if (i >= Btok) return;
    int s = ts[i], d = td[i];
    if (s == d) {
        int p = atomicAdd(&counts[s], 1);
        lists[(size_t)s * Btok + p] = (unsigned)i | SUMF | DIFFF;
    } else {
        int p = atomicAdd(&counts[s], 1);
        lists[(size_t)s * Btok + p] = (unsigned)i | SUMF;
        int q = atomicAdd(&counts[d], 1);
        lists[(size_t)d * Btok + q] = (unsigned)i | DIFFF;
    }
}

// ---------------------------------------------------------------------------
// Routed expert FF1: for tile t (blockIdx.z), gathered rows m0..m0+127,
// N-chunk blockIdx.x (128 of 2048). Epilogue folds gelu(+be1)*v_sum/v_diff,
// half-wave reduces over N, writes per-(token,chunk) partials.
// ---------------------------------------------------------------------------
__global__ __launch_bounds__(256) void expert_ff1_kernel(
    const unsigned short* __restrict__ X,     // B x 1024 bf16
    const unsigned short* __restrict__ We1b,  // 8 x 1024 x 2048 bf16
    const float* __restrict__ be1,            // 8 x 2048
    const float* __restrict__ vsum,           // 8 x 2048
    const float* __restrict__ vdiff,          // 8 x 2048
    const unsigned int* __restrict__ lists,   // 8 x B
    const int* __restrict__ counts,           // 8
    float* __restrict__ psum,                 // B x 32
    float* __restrict__ pdiff,                // B x 32
    int Btok) {
    const int t = blockIdx.z;
    const int cnt = counts[t];
    const int m0 = blockIdx.y * 128;
    if (m0 >= cnt) return;
    const int n0 = blockIdx.x * 128;
    const int K = 1024, N = 2048;
    const unsigned short* Wt = We1b + (size_t)t * K * N;

    const int tid  = threadIdx.x;
    const int lane = tid & 31;
    const int wv   = tid >> 5;
    const int wave_m = wv >> 1, wave_n = wv & 1;
    const int wm0 = wave_m * 32, wn0 = wave_n * 64;

    __shared__ __align__(16) unsigned short As[2][128][40];
    __shared__ __align__(16) unsigned short Bs[2][32][136];
    __shared__ unsigned int Ent[128];

    if (tid < 128) {
        int r = m0 + tid;
        Ent[tid] = (r < cnt) ? lists[(size_t)t * Btok + r] : 0u;
    }
    __syncthreads();

    const int ra0 = tid >> 2,         ka0 = (tid & 3) * 8;
    const int ra1 = (tid + 256) >> 2, ka1 = ((tid + 256) & 3) * 8;
    const int kb0 = tid >> 4,         nb0 = (tid & 15) * 8;
    const int kb1 = (tid + 256) >> 4, nb1 = ((tid + 256) & 15) * 8;
    const unsigned short* aP0 = X + (size_t)(Ent[ra0] & TOKM) * K + ka0;
    const unsigned short* aP1 = X + (size_t)(Ent[ra1] & TOKM) * K + ka1;
    const unsigned short* bP0 = Wt + (size_t)kb0 * N + n0 + nb0;
    const unsigned short* bP1 = Wt + (size_t)kb1 * N + n0 + nb1;

    v8f acc[2][4];
    const v8f vz = {0.f, 0.f, 0.f, 0.f, 0.f, 0.f, 0.f, 0.f};
    #pragma unroll
    for (int m = 0; m < 2; ++m)
        #pragma unroll
        for (int n = 0; n < 4; ++n) acc[m][n] = vz;

    const int mrow = wm0 + (lane & 15);
    const int kba  = (lane >> 4) * 8;

    cp16(aP0, &As[0][ra0][ka0]);
    cp16(aP1, &As[0][ra1][ka1]);
    cp16(bP0, &Bs[0][kb0][nb0]);
    cp16(bP1, &Bs[0][kb1][nb1]);
    cp_commit();
    __syncthreads();

    int cur = 0;
    for (int k0 = 0; k0 < K; k0 += 32) {
        const int nxt = cur ^ 1;
        if (k0 + 32 < K) {
            cp16(aP0 + k0 + 32, &As[nxt][ra0][ka0]);
            cp16(aP1 + k0 + 32, &As[nxt][ra1][ka1]);
            cp16(bP0 + (size_t)(k0 + 32) * N, &Bs[nxt][kb0][nb0]);
            cp16(bP1 + (size_t)(k0 + 32) * N, &Bs[nxt][kb1][nb1]);
        }

        Frag16 af[2], bf[4];
        #pragma unroll
        for (int m = 0; m < 2; ++m) {
            af[m].u[0] = *reinterpret_cast<const u32x4*>(&As[cur][mrow + m * 16][kba]);
            af[m].u[1] = *reinterpret_cast<const u32x4*>(&As[cur][mrow + m * 16][kba + 16]);
        }
        #pragma unroll
        for (int n = 0; n < 4; ++n) {
            int nc = wn0 + n * 16 + (lane & 15);
            #pragma unroll
            for (int j = 0; j < 4; ++j) {
                bf[n].u[0][j] = (unsigned)Bs[cur][kba + 2 * j][nc] | ((unsigned)Bs[cur][kba + 2 * j + 1][nc] << 16);
                bf[n].u[1][j] = (unsigned)Bs[cur][kba + 16 + 2 * j][nc] | ((unsigned)Bs[cur][kba + 17 + 2 * j][nc] << 16);
            }
        }
        #pragma unroll
        for (int m = 0; m < 2; ++m)
            #pragma unroll
            for (int n = 0; n < 4; ++n)
                acc[m][n] = __builtin_amdgcn_wmma_f32_16x16x32_bf16(
                    false, af[m].bf, false, bf[n].bf, (short)0, acc[m][n],
                    false, false);

        cp_commit();
        __syncthreads();
        cur = nxt;
    }

    // epilogue: gelu(+be1) dotted with v_sum/v_diff over this wave's 64 cols
    float bcol[4], vs4[4], vd4[4];
    #pragma unroll
    for (int n = 0; n < 4; ++n) {
        int col = n0 + wn0 + n * 16 + (lane & 15);
        size_t gi = (size_t)t * 2048 + col;
        bcol[n] = be1[gi];
        vs4[n]  = vsum[gi];
        vd4[n]  = vdiff[gi];
    }
    const int chunk = blockIdx.x * 2 + wave_n;   // 0..31
    #pragma unroll
    for (int m = 0; m < 2; ++m) {
        #pragma unroll
        for (int i = 0; i < 8; ++i) {
            float ps = 0.f, pd = 0.f;
            #pragma unroll
            for (int n = 0; n < 4; ++n) {
                float g = gelu_exact(acc[m][n][i] + bcol[n]);
                ps += g * vs4[n];
                pd += g * vd4[n];
            }
            for (int o = 1; o < 16; o <<= 1) {
                ps += __shfl_xor(ps, o, 16);
                pd += __shfl_xor(pd, o, 16);
            }
            if ((lane & 15) == 0) {
                int rloc = wm0 + m * 16 + i + ((lane >> 4) << 3);
                if (m0 + rloc < cnt) {
                    unsigned e = Ent[rloc];
                    unsigned tok = e & TOKM;
                    if (e & SUMF)  psum[(size_t)tok * 32 + chunk] = ps;
                    if (e & DIFFF) pdiff[(size_t)tok * 32 + chunk] = pd;
                }
            }
        }
    }
}

// out[i] = bs + c_sum[ts[i]] + sum(partials); out[B+i] likewise.
__global__ void finalize_kernel(const float* __restrict__ psum,
                                const float* __restrict__ pdiff,
                                const int* __restrict__ ts,
                                const int* __restrict__ td,
                                const float* __restrict__ cs,
                                const float* __restrict__ cd,
                                const float* __restrict__ bs,
                                const float* __restrict__ bd,
                                float* __restrict__ out, int Btok) {
    int i = blockIdx.x * 256 + threadIdx.x;
    if (i >= Btok) return;
    float s = bs[0] + cs[ts[i]];
    float d = bd[0] + cd[td[i]];
    const float* p = psum + (size_t)i * 32;
    const float* q = pdiff + (size_t)i * 32;
    #pragma unroll
    for (int c = 0; c < 32; ++c) { s += p[c]; d += q[c]; }
    out[i] = s;
    out[Btok + i] = d;
}

// ---------------------------------------------------------------------------
extern "C" void kernel_launch(void* const* d_in, const int* in_sizes, int n_in,
                              void* d_out, int out_size, void* d_ws,
                              size_t ws_size, hipStream_t stream) {
    const int B = in_sizes[0];        // 65536
    const int D = 1024, T = 8, N2 = 2048;

    const float* a       = (const float*)d_in[0];
    const float* b       = (const float*)d_in[1];
    const float* W_in    = (const float*)d_in[2];
    const float* b_in    = (const float*)d_in[3];
    const float* ln_g    = (const float*)d_in[4];
    const float* ln_b    = (const float*)d_in[5];
    const float* Wr1_s   = (const float*)d_in[6];
    const float* br1_s   = (const float*)d_in[7];
    const float* Wr2_s   = (const float*)d_in[8];
    const float* br2_s   = (const float*)d_in[9];
    const float* Wr1_d   = (const float*)d_in[10];
    const float* br1_d   = (const float*)d_in[11];
    const float* Wr2_d   = (const float*)d_in[12];
    const float* br2_d   = (const float*)d_in[13];
    const float* We1     = (const float*)d_in[14];
    const float* be1     = (const float*)d_in[15];
    const float* We2     = (const float*)d_in[16];
    const float* be2     = (const float*)d_in[17];
    const float* w_sum   = (const float*)d_in[18];
    const float* bs      = (const float*)d_in[19];
    const float* w_diff  = (const float*)d_in[20];
    const float* bd      = (const float*)d_in[21];
    float* out = (float*)d_out;

    // workspace carve (256B aligned)
    char* w = (char*)d_ws;
    auto carve = [&](size_t bytes) {
        void* p = (void*)w;
        w += (bytes + 255) & ~(size_t)255;
        return p;
    };
    unsigned short* xbf   = (unsigned short*)carve((size_t)B * D * 2);
    unsigned short* hbf   = (unsigned short*)carve((size_t)B * D * 2);
    unsigned short* wr1sb = (unsigned short*)carve((size_t)D * D * 2);
    unsigned short* wr1db = (unsigned short*)carve((size_t)D * D * 2);
    unsigned short* we1b  = (unsigned short*)carve((size_t)T * D * N2 * 2);
    float* vsum  = (float*)carve((size_t)T * N2 * 4);
    float* vdiff = (float*)carve((size_t)T * N2 * 4);
    float* csum  = (float*)carve(T * 4);
    float* cdiff = (float*)carve(T * 4);
    int* tiles_s = (int*)carve((size_t)B * 4);
    int* tiles_d = (int*)carve((size_t)B * 4);
    int* counts  = (int*)carve(256);
    unsigned int* lists = (unsigned int*)carve((size_t)T * B * 4);
    float* psum  = (float*)carve((size_t)B * 32 * 4);
    float* pdiff = (float*)carve((size_t)B * 32 * 4);

    // 1. weight precompute
    cvt_bf16_kernel<<<(D * D) / 1024, 256, 0, stream>>>(Wr1_s, wr1sb, D * D);
    cvt_bf16_kernel<<<(D * D) / 1024, 256, 0, stream>>>(Wr1_d, wr1db, D * D);
    cvt_bf16_kernel<<<(T * D * N2) / 1024, 256, 0, stream>>>(We1, we1b, T * D * N2);
    vproj_kernel<<<(T * N2) / 8, 256, 0, stream>>>(We2, w_sum, w_diff, vsum, vdiff);
    cvec_kernel<<<T, 256, 0, stream>>>(be2, w_sum, w_diff, csum, cdiff);

    // 2. encoder
    encode_kernel<<<B, 256, 0, stream>>>(a, b, W_in, b_in, ln_g, ln_b, xbf);

    // 3. routers (WMMA GEMM + argmax), sequential reuse of hbf
    dim3 gg(D / 128, B / 128);
    gemm_gelu_kernel<<<gg, 256, 0, stream>>>(xbf, wr1sb, br1_s, hbf, B, D, D);
    router_argmax_kernel<<<B / 8, 256, 0, stream>>>(hbf, Wr2_s, br2_s, tiles_s);
    gemm_gelu_kernel<<<gg, 256, 0, stream>>>(xbf, wr1db, br1_d, hbf, B, D, D);
    router_argmax_kernel<<<B / 8, 256, 0, stream>>>(hbf, Wr2_d, br2_d, tiles_d);

    // 4. dispatch
    zero_counts_kernel<<<1, 32, 0, stream>>>(counts);
    dispatch_kernel<<<B / 256, 256, 0, stream>>>(tiles_s, tiles_d, counts, lists, B);

    // 5. routed expert FF1 (FF2 collapsed into vsum/vdiff epilogue)
    dim3 ge(N2 / 128, B / 128, T);
    expert_ff1_kernel<<<ge, 256, 0, stream>>>(xbf, we1b, be1, vsum, vdiff,
                                              lists, counts, psum, pdiff, B);

    // 6. finalize
    finalize_kernel<<<B / 256, 256, 0, stream>>>(psum, pdiff, tiles_s, tiles_d,
                                                 csum, cdiff, bs, bd, out, B);
}
